// ImprovedSeq2SeqModel_65034394796346
// MI455X (gfx1250) — compile-verified, matrix operand order checked by reading, not run on hardware
//
#include <hip/hip_runtime.h>
#include <hip/hip_bf16.h>

// ---------------------------------------------------------------------------
// Seq2Seq LSTM + attention for MI455X (gfx1250, wave32, WMMA).
//
//  * LSTM recurrence is batch-row independent -> persistent workgroups, one
//    16-row batch tile each (WMMA M=16), zero inter-WG synchronization.
//  * All matmuls lowered to v_wmma_f32_16x16x32_bf16 (fp32 accumulate),
//    weights pre-packed to bf16 K-major (K padded to multiple of 32).
//  * h state: bf16 in LDS (A-fragments re-read per step, ds_load_b128);
//    c state: registers in WMMA C/D lane layout (never leaves the wave).
//  * Weights stream from L2 each step (~4MB total bf16, fits 192MB L2).
//  * Pure global->LDS staging copies use the CDNA5 async engine
//    (global_load_async_to_lds_b128 + s_wait_asynccnt) instead of going
//    through VGPRs: no register traffic, tracked on ASYNCcnt.
// ---------------------------------------------------------------------------

#define S_LEN 168
#define B_SZ  512
#define HID   256
#define TDEC  24

typedef __attribute__((ext_vector_type(16))) __bf16 v16bf;
typedef __attribute__((ext_vector_type(8)))  float  v8f;

union Frag { v16bf v; uint4 q[2]; unsigned short u[16]; };

__device__ __forceinline__ unsigned short f2bf(float f) {
  union { float f; unsigned u; } c; c.f = f;
  unsigned r = c.u + 0x7FFFu + ((c.u >> 16) & 1u);   // round-to-nearest-even
  return (unsigned short)(r >> 16);
}
__device__ __forceinline__ float bf2f(unsigned short h) {
  union { unsigned u; float f; } c; c.u = ((unsigned)h) << 16;
  return c.f;
}
__device__ __forceinline__ float sigm(float x) { return 1.0f / (1.0f + __expf(-x)); }

__device__ __forceinline__ v8f splat8(float v) {
  v8f x;
  #pragma unroll
  for (int i = 0; i < 8; ++i) x[i] = v;
  return x;
}

__device__ __forceinline__ v8f wmma_bf16(const Frag& a, const Frag& b, v8f c) {
  // 8 args: (neg_a, A, neg_b, B, c_mod, C, reuse_a, reuse_b)
  return __builtin_amdgcn_wmma_f32_16x16x32_bf16(false, a.v, false, b.v,
                                                 (short)0, c, false, false);
}

// Async DMA: 16 bytes global -> LDS, no VGPR data path, tracked on ASYNCcnt.
// lds_addr: 32-bit LDS byte address (low 32 bits of the generic pointer).
__device__ __forceinline__ void async_copy_b128(unsigned lds_addr, const void* gptr) {
  asm volatile("global_load_async_to_lds_b128 %0, %1, off"
               :: "v"(lds_addr), "v"(gptr)
               : "memory");
}
__device__ __forceinline__ void wait_async0() {
  asm volatile("s_wait_asynccnt 0x0" ::: "memory");
}

// A-fragment (16xK tile, row-major in LDS, 16-bit elems):
// lane L in [0,16): row M=L, kb=0; lane L in [16,32): row M=L-16, kb=8
// (per ISA 7.12.2 16-bit A layout)
__device__ __forceinline__ Frag frag_a_lds(const unsigned short* base, int stride,
                                           int k0, int lane) {
  Frag f;
  int row = lane & 15;
  int kb  = (lane >> 4) << 3;
  f.q[0] = *(const uint4*)(base + row * stride + k0 + kb);
  f.q[1] = *(const uint4*)(base + row * stride + k0 + kb + 16);
  return f;
}

// B-fragment (Kx16 tile). Weights packed as W[n][k] bf16 with row stride kpad,
// so per-lane k-runs of 8 are contiguous 16-byte loads.
__device__ __forceinline__ Frag frag_b_glb(const unsigned short* w, int kpad,
                                           int ncol, int k0, int lane) {
  Frag f;
  const unsigned short* p =
      w + (size_t)(ncol + (lane & 15)) * kpad + k0 + ((lane >> 4) << 3);
  f.q[0] = *(const uint4*)p;
  f.q[1] = *(const uint4*)(p + 16);
  return f;
}

// ---------------------------------------------------------------------------
// Weight packing: dst[n*Kpad + k] = bf16(src[n*srcStride + srcOff + k]), 0-pad.
// ---------------------------------------------------------------------------
__global__ void pack_weight_kernel(const float* __restrict__ src,
                                   unsigned short* __restrict__ dst,
                                   int N, int K, int Kpad, int srcStride, int srcOff) {
  int idx = blockIdx.x * blockDim.x + threadIdx.x;
  if (idx >= N * Kpad) return;
  int n = idx / Kpad, k = idx - n * Kpad;
  dst[idx] = (k < K) ? f2bf(src[(size_t)n * srcStride + srcOff + k])
                     : (unsigned short)0;
}

// ---------------------------------------------------------------------------
// Persistent LSTM layer. Grid = 32 blocks (B/16), 256 threads = 8 waves.
// Wave w owns hidden columns [w*32, w*32+32): 2 N-tiles per gate, 4 gates
// -> 8 WMMA accumulators. XK = padded input width (32 for layer0, 256 else).
// ---------------------------------------------------------------------------
template <int XK, bool XF32>
__global__ __launch_bounds__(256) void lstm_layer_kernel(
    const void* __restrict__ xin,            // XF32: float src[B][S][16]; else ushort y[S][B][256]
    const unsigned short* __restrict__ wih,  // [1024][XK]   bf16 packed
    const unsigned short* __restrict__ whh,  // [1024][256]  bf16 packed
    const float* __restrict__ bias,          // [1024]
    unsigned short* __restrict__ yout,       // [S][B][256]  bf16
    float* __restrict__ hfin,                // [B][256] final h (fp32)
    float* __restrict__ cfin)                // [B][256] final c (fp32)
{
  __shared__ unsigned short h_lds[16 * HID];
  __shared__ unsigned short x_lds[16 * XK];

  const int tid  = threadIdx.x;
  const int lane = tid & 31;
  const int wv   = tid >> 5;            // 0..7
  const int b0   = blockIdx.x * 16;
  const int hc   = wv * 32;
  const int mb   = (lane >> 4) << 3;    // row base for C/D fragments
  const int nl   = lane & 15;

  for (int i = tid; i < 16 * HID; i += 256) h_lds[i] = 0;
  for (int i = tid; i < 16 * XK;  i += 256) x_lds[i] = 0;

  float creg[2][8];
  #pragma unroll
  for (int j = 0; j < 2; ++j)
    #pragma unroll
    for (int r = 0; r < 8; ++r) creg[j][r] = 0.f;

  float bv[4][2];
  #pragma unroll
  for (int g = 0; g < 4; ++g)
    #pragma unroll
    for (int j = 0; j < 2; ++j)
      bv[g][j] = bias[g * HID + hc + j * 16 + nl];

  const unsigned x_lds_base = (unsigned)(size_t)&x_lds[0];
  __syncthreads();

  for (int t = 0; t < S_LEN; ++t) {
    // ---- stage x tile into LDS (bf16) ----
    if (XF32) {
      const float* src = (const float*)xin;
      int row = tid >> 4, col = tid & 15;   // 16x16 payload, cols 16..31 stay 0
      x_lds[row * XK + col] =
          f2bf(src[(size_t)(b0 + row) * (S_LEN * 16) + (size_t)t * 16 + col]);
    } else {
      // raw bf16 bytes -> async DMA straight into LDS (2 x b128 per thread)
      const unsigned short* xg = (const unsigned short*)xin +
          (size_t)t * B_SZ * HID + (size_t)b0 * HID;
      #pragma unroll
      for (int i = 0; i < 2; ++i) {
        int e = (tid + i * 256) * 8;        // ushort index, 8 ushorts = 16B
        async_copy_b128(x_lds_base + (unsigned)(e * 2), xg + e);
      }
      wait_async0();
    }
    __syncthreads();

    v8f acc[4][2];
    #pragma unroll
    for (int g = 0; g < 4; ++g)
      #pragma unroll
      for (int j = 0; j < 2; ++j) acc[g][j] = splat8(bv[g][j]);

    // ---- x @ W_ih^T ----
    #pragma unroll
    for (int kk = 0; kk < XK / 32; ++kk) {
      Frag a = frag_a_lds(x_lds, XK, kk * 32, lane);
      #pragma unroll
      for (int g = 0; g < 4; ++g)
        #pragma unroll
        for (int j = 0; j < 2; ++j)
          acc[g][j] = wmma_bf16(a, frag_b_glb(wih, XK, g * HID + hc + j * 16, kk * 32, lane), acc[g][j]);
    }
    // ---- h @ W_hh^T ----
    #pragma unroll
    for (int kk = 0; kk < HID / 32; ++kk) {
      Frag a = frag_a_lds(h_lds, HID, kk * 32, lane);
      #pragma unroll
      for (int g = 0; g < 4; ++g)
        #pragma unroll
        for (int j = 0; j < 2; ++j)
          acc[g][j] = wmma_bf16(a, frag_b_glb(whh, HID, g * HID + hc + j * 16, kk * 32, lane), acc[g][j]);
    }
    __syncthreads();   // everyone done reading h_lds / x_lds

    // ---- elementwise cell update; h -> LDS + global, c stays in regs ----
    #pragma unroll
    for (int j = 0; j < 2; ++j) {
      int col = hc + j * 16 + nl;
      #pragma unroll
      for (int r = 0; r < 8; ++r) {
        float iv = acc[0][j][r], fv = acc[1][j][r];
        float gv = acc[2][j][r], ov = acc[3][j][r];
        float cn = sigm(fv) * creg[j][r] + sigm(iv) * tanhf(gv);
        float hn = sigm(ov) * tanhf(cn);
        creg[j][r] = cn;
        int m = mb + r;
        unsigned short hb = f2bf(hn);
        h_lds[m * HID + col] = hb;
        yout[(size_t)t * B_SZ * HID + (size_t)(b0 + m) * HID + col] = hb;
        if (t == S_LEN - 1) {
          hfin[(size_t)(b0 + m) * HID + col] = hn;
          cfin[(size_t)(b0 + m) * HID + col] = cn;
        }
      }
    }
    __syncthreads();
  }
}

// ---------------------------------------------------------------------------
// enc_proj = enc_out @ We^T : M=S*B (tiled 16), K=256, N=256.
// Grid = 168*32 blocks, 512 threads = 16 waves (one N-tile each).
// Output layout [b][s][g] fp32 for decoder locality.
// ---------------------------------------------------------------------------
__global__ __launch_bounds__(512) void encproj_kernel(
    const unsigned short* __restrict__ enc_out,  // [S][B][256] bf16
    const unsigned short* __restrict__ wep,      // [256][256] bf16 packed
    float* __restrict__ encproj)                 // [B][S][256] fp32
{
  __shared__ unsigned short a_lds[16 * HID];
  const int tid  = threadIdx.x;
  const int lane = tid & 31;
  const int wv   = tid >> 5;             // 0..15
  const int t    = blockIdx.x >> 5;
  const int b0   = (blockIdx.x & 31) * 16;

  // async DMA the 8KB A-tile into LDS (one b128 per thread)
  {
    const unsigned short* src =
        enc_out + (size_t)t * B_SZ * HID + (size_t)b0 * HID;
    unsigned lbase = (unsigned)(size_t)&a_lds[0];
    int e = tid * 8;                      // 8 ushorts = 16B
    async_copy_b128(lbase + (unsigned)(e * 2), src + e);
    wait_async0();
  }
  __syncthreads();

  v8f acc = splat8(0.f);
  #pragma unroll
  for (int kk = 0; kk < HID / 32; ++kk) {
    Frag a = frag_a_lds(a_lds, HID, kk * 32, lane);
    Frag b = frag_b_glb(wep, HID, wv * 16, kk * 32, lane);
    acc = wmma_bf16(a, b, acc);
  }
  int n  = wv * 16 + (lane & 15);
  int mb = (lane >> 4) << 3;
  #pragma unroll
  for (int r = 0; r < 8; ++r) {
    int m = mb + r;
    encproj[(size_t)(b0 + m) * (S_LEN * HID) + (size_t)t * HID + n] = acc[r];
  }
}

// ---------------------------------------------------------------------------
// Decoder: persistent per-batch-tile; T=24 steps of attention + 2 LSTM cells.
// Grid = 32 blocks, 256 threads = 8 waves.
// Note: pos_bias*(t/24) adds the same constant to every s -> cancels in
// softmax, so it is elided from the score computation.
// ---------------------------------------------------------------------------
__global__ __launch_bounds__(256) void decoder_kernel(
    const float* __restrict__ src,
    const unsigned short* __restrict__ enc_out,   // [S][B][256] bf16
    const float* __restrict__ encproj,            // [B][S][256] fp32
    const unsigned short* __restrict__ wd_p,      // [256][256]
    const float* __restrict__ attn_b,
    const float* __restrict__ attn_v,
    const unsigned short* __restrict__ dih0,      // [1024][288]
    const unsigned short* __restrict__ dhh0,      // [1024][256]
    const float* __restrict__ db0,
    const unsigned short* __restrict__ dih1,      // [1024][256]
    const unsigned short* __restrict__ dhh1,      // [1024][256]
    const float* __restrict__ db1,
    const float* __restrict__ fc_w,
    const float* __restrict__ fc_b,
    const float* __restrict__ h0e, const float* __restrict__ c0e,
    const float* __restrict__ h1e, const float* __restrict__ c1e,
    float* __restrict__ out)                      // [B][T] fp32
{
  __shared__ unsigned short h0_lds[16 * HID];
  __shared__ unsigned short h1_lds[16 * HID];
  __shared__ unsigned short x0_lds[16 * 288];
  __shared__ float q_lds[16 * HID];
  __shared__ float sc_lds[16 * S_LEN];
  __shared__ float pred_lds[16];
  __shared__ float din_lds[16];

  const int tid  = threadIdx.x;
  const int lane = tid & 31;
  const int wv   = tid >> 5;
  const int b0   = blockIdx.x * 16;
  const int hc   = wv * 32;
  const int mb   = (lane >> 4) << 3;
  const int nl   = lane & 15;

  for (int i = tid; i < 16 * HID; i += 256) {
    h0_lds[i] = f2bf(h0e[(size_t)b0 * HID + i]);
    h1_lds[i] = f2bf(h1e[(size_t)b0 * HID + i]);
  }
  for (int i = tid; i < 16 * 288; i += 256) x0_lds[i] = 0;
  if (tid < 16)
    din_lds[tid] = src[(size_t)(b0 + tid) * (S_LEN * 16) + (S_LEN - 1) * 16 + 15];

  float c0r[2][8], c1r[2][8];
  float qbv[2], fcw[2], b0v[4][2], b1v[4][2];
  #pragma unroll
  for (int j = 0; j < 2; ++j) {
    int col = hc + j * 16 + nl;
    qbv[j] = attn_b[col];
    fcw[j] = fc_w[col];
    #pragma unroll
    for (int r = 0; r < 8; ++r) {
      int m = mb + r;
      c0r[j][r] = c0e[(size_t)(b0 + m) * HID + col];
      c1r[j][r] = c1e[(size_t)(b0 + m) * HID + col];
    }
    #pragma unroll
    for (int g = 0; g < 4; ++g) {
      b0v[g][j] = db0[g * HID + col];
      b1v[g][j] = db1[g * HID + col];
    }
  }
  float vreg[8];
  #pragma unroll
  for (int i = 0; i < 8; ++i) vreg[i] = attn_v[lane * 8 + i];
  const float fcb = fc_b[0];
  __syncthreads();

  for (int t = 0; t < TDEC; ++t) {
    // ---- phase 1: q = h1 @ Wd^T + attn_b ----
    v8f qa[2];
    #pragma unroll
    for (int j = 0; j < 2; ++j) qa[j] = splat8(qbv[j]);
    #pragma unroll
    for (int kk = 0; kk < HID / 32; ++kk) {
      Frag a = frag_a_lds(h1_lds, HID, kk * 32, lane);
      #pragma unroll
      for (int j = 0; j < 2; ++j)
        qa[j] = wmma_bf16(a, frag_b_glb(wd_p, HID, hc + j * 16, kk * 32, lane), qa[j]);
    }
    #pragma unroll
    for (int j = 0; j < 2; ++j)
      #pragma unroll
      for (int r = 0; r < 8; ++r)
        q_lds[(mb + r) * HID + hc + j * 16 + nl] = qa[j][r];
    __syncthreads();

    // ---- phase 2: scores[b][s] = sum_h tanh(ep+q)*v  (wave w: rows 2w,2w+1)
    #pragma unroll
    for (int bi = 0; bi < 2; ++bi) {
      int b = wv * 2 + bi;
      const float* ep = encproj + (size_t)(b0 + b) * (S_LEN * HID);
      float q8[8];
      #pragma unroll
      for (int i = 0; i < 8; ++i) q8[i] = q_lds[b * HID + lane * 8 + i];
      for (int s = 0; s < S_LEN; ++s) {
        const float* e = ep + (size_t)s * HID + lane * 8;
        float part = 0.f;
        #pragma unroll
        for (int i = 0; i < 8; ++i) part += tanhf(e[i] + q8[i]) * vreg[i];
        #pragma unroll
        for (int off = 16; off > 0; off >>= 1) part += __shfl_down(part, off, 32);
        if (lane == 0) sc_lds[b * S_LEN + s] = part;
      }
    }
    __syncthreads();

    // ---- phase 3: softmax over s (per row), zero pred accumulators ----
    #pragma unroll
    for (int bi = 0; bi < 2; ++bi) {
      int b = wv * 2 + bi;
      float mx = -3.0e30f;
      for (int s = lane; s < S_LEN; s += 32) mx = fmaxf(mx, sc_lds[b * S_LEN + s]);
      #pragma unroll
      for (int off = 16; off > 0; off >>= 1) mx = fmaxf(mx, __shfl_xor(mx, off, 32));
      float sum = 0.f;
      for (int s = lane; s < S_LEN; s += 32) {
        float e = __expf(sc_lds[b * S_LEN + s] - mx);
        sc_lds[b * S_LEN + s] = e;
        sum += e;
      }
      #pragma unroll
      for (int off = 16; off > 0; off >>= 1) sum += __shfl_xor(sum, off, 32);
      float inv = 1.0f / sum;
      for (int s = lane; s < S_LEN; s += 32) sc_lds[b * S_LEN + s] *= inv;
    }
    if (tid < 16) pred_lds[tid] = 0.f;
    __syncthreads();

    // ---- phase 4: ctx[b][h] = sum_s w[b][s]*enc_out[s][b][h]; build x0 ----
    for (int idx = tid; idx < 16 * HID; idx += 256) {
      int b = idx >> 8, h = idx & 255;
      const unsigned short* eo = enc_out + (size_t)(b0 + b) * HID + h;
      float acc = 0.f;
      for (int s = 0; s < S_LEN; ++s) {
        if (s + 8 < S_LEN)
          __builtin_prefetch(eo + (size_t)(s + 8) * B_SZ * HID, 0, 0);
        acc += sc_lds[b * S_LEN + s] * bf2f(eo[(size_t)s * B_SZ * HID]);
      }
      x0_lds[b * 288 + 1 + h] = f2bf(acc);
    }
    if (tid < 16) x0_lds[tid * 288] = f2bf(din_lds[tid]);
    __syncthreads();

    // ---- phase 5: cell0: gates = x0@W_ih0^T + h0@W_hh0^T + b0 ----
    v8f acc0[4][2];
    #pragma unroll
    for (int g = 0; g < 4; ++g)
      #pragma unroll
      for (int j = 0; j < 2; ++j) acc0[g][j] = splat8(b0v[g][j]);
    #pragma unroll
    for (int kk = 0; kk < 288 / 32; ++kk) {
      Frag a = frag_a_lds(x0_lds, 288, kk * 32, lane);
      #pragma unroll
      for (int g = 0; g < 4; ++g)
        #pragma unroll
        for (int j = 0; j < 2; ++j)
          acc0[g][j] = wmma_bf16(a, frag_b_glb(dih0, 288, g * HID + hc + j * 16, kk * 32, lane), acc0[g][j]);
    }
    #pragma unroll
    for (int kk = 0; kk < HID / 32; ++kk) {
      Frag a = frag_a_lds(h0_lds, HID, kk * 32, lane);
      #pragma unroll
      for (int g = 0; g < 4; ++g)
        #pragma unroll
        for (int j = 0; j < 2; ++j)
          acc0[g][j] = wmma_bf16(a, frag_b_glb(dhh0, HID, g * HID + hc + j * 16, kk * 32, lane), acc0[g][j]);
    }
    __syncthreads();
    #pragma unroll
    for (int j = 0; j < 2; ++j) {
      int col = hc + j * 16 + nl;
      #pragma unroll
      for (int r = 0; r < 8; ++r) {
        float cn = sigm(acc0[1][j][r]) * c0r[j][r] +
                   sigm(acc0[0][j][r]) * tanhf(acc0[2][j][r]);
        float hn = sigm(acc0[3][j][r]) * tanhf(cn);
        c0r[j][r] = cn;
        h0_lds[(mb + r) * HID + col] = f2bf(hn);
      }
    }
    __syncthreads();

    // ---- phase 6: cell1: gates = h0n@W_ih1^T + h1@W_hh1^T + b1 ----
    v8f acc1[4][2];
    #pragma unroll
    for (int g = 0; g < 4; ++g)
      #pragma unroll
      for (int j = 0; j < 2; ++j) acc1[g][j] = splat8(b1v[g][j]);
    #pragma unroll
    for (int kk = 0; kk < HID / 32; ++kk) {
      Frag a = frag_a_lds(h0_lds, HID, kk * 32, lane);
      #pragma unroll
      for (int g = 0; g < 4; ++g)
        #pragma unroll
        for (int j = 0; j < 2; ++j)
          acc1[g][j] = wmma_bf16(a, frag_b_glb(dih1, HID, g * HID + hc + j * 16, kk * 32, lane), acc1[g][j]);
    }
    #pragma unroll
    for (int kk = 0; kk < HID / 32; ++kk) {
      Frag a = frag_a_lds(h1_lds, HID, kk * 32, lane);
      #pragma unroll
      for (int g = 0; g < 4; ++g)
        #pragma unroll
        for (int j = 0; j < 2; ++j)
          acc1[g][j] = wmma_bf16(a, frag_b_glb(dhh1, HID, g * HID + hc + j * 16, kk * 32, lane), acc1[g][j]);
    }
    __syncthreads();
    #pragma unroll
    for (int j = 0; j < 2; ++j) {
      int col = hc + j * 16 + nl;
      float fw = fcw[j];
      #pragma unroll
      for (int r = 0; r < 8; ++r) {
        float cn = sigm(acc1[1][j][r]) * c1r[j][r] +
                   sigm(acc1[0][j][r]) * tanhf(acc1[2][j][r]);
        float hn = sigm(acc1[3][j][r]) * tanhf(cn);
        c1r[j][r] = cn;
        h1_lds[(mb + r) * HID + col] = f2bf(hn);
        atomicAdd(&pred_lds[mb + r], hn * fw);   // ds_add_f32
      }
    }
    __syncthreads();

    // ---- phase 7: write prediction, feed back as next dec_in ----
    if (tid < 16) {
      float p = pred_lds[tid] + fcb;
      out[(size_t)(b0 + tid) * TDEC + t] = p;
      din_lds[tid] = p;
    }
    __syncthreads();
  }
}

// ---------------------------------------------------------------------------
// Host-side orchestration
// ---------------------------------------------------------------------------
extern "C" void kernel_launch(void* const* d_in, const int* in_sizes, int n_in,
                              void* d_out, int out_size, void* d_ws, size_t ws_size,
                              hipStream_t stream) {
  const float* src      = (const float*)d_in[0];
  const float* e_wih0   = (const float*)d_in[1];
  const float* e_whh0   = (const float*)d_in[2];
  const float* e_b0     = (const float*)d_in[3];
  const float* e_wih1   = (const float*)d_in[4];
  const float* e_whh1   = (const float*)d_in[5];
  const float* e_b1     = (const float*)d_in[6];
  const float* d_wih0   = (const float*)d_in[7];
  const float* d_whh0   = (const float*)d_in[8];
  const float* d_b0     = (const float*)d_in[9];
  const float* d_wih1   = (const float*)d_in[10];
  const float* d_whh1   = (const float*)d_in[11];
  const float* d_b1     = (const float*)d_in[12];
  const float* attn_w   = (const float*)d_in[13];
  const float* attn_b   = (const float*)d_in[14];
  const float* attn_v   = (const float*)d_in[15];
  // d_in[16] = pos_bias: elided (constant-per-s shift cancels in softmax)
  const float* fc_w     = (const float*)d_in[17];
  const float* fc_b     = (const float*)d_in[18];
  float* out = (float*)d_out;

  char* ws = (char*)d_ws;
  size_t off = 0;
  auto alloc = [&](size_t bytes) -> void* {
    void* p = ws + off;
    off = (off + bytes + 255) & ~(size_t)255;
    return p;
  };
  unsigned short* p_eih0 = (unsigned short*)alloc(1024u * 32  * 2);
  unsigned short* p_ehh0 = (unsigned short*)alloc(1024u * 256 * 2);
  unsigned short* p_eih1 = (unsigned short*)alloc(1024u * 256 * 2);
  unsigned short* p_ehh1 = (unsigned short*)alloc(1024u * 256 * 2);
  unsigned short* p_dih0 = (unsigned short*)alloc(1024u * 288 * 2);
  unsigned short* p_dhh0 = (unsigned short*)alloc(1024u * 256 * 2);
  unsigned short* p_dih1 = (unsigned short*)alloc(1024u * 256 * 2);
  unsigned short* p_dhh1 = (unsigned short*)alloc(1024u * 256 * 2);
  unsigned short* p_we   = (unsigned short*)alloc(256u  * 256 * 2);
  unsigned short* p_wd   = (unsigned short*)alloc(256u  * 256 * 2);
  unsigned short* y0     = (unsigned short*)alloc((size_t)S_LEN * B_SZ * HID * 2);
  unsigned short* eob    = (unsigned short*)alloc((size_t)S_LEN * B_SZ * HID * 2);
  float* epj             = (float*)alloc((size_t)B_SZ * S_LEN * HID * 4);
  float* h0e             = (float*)alloc((size_t)B_SZ * HID * 4);
  float* c0e             = (float*)alloc((size_t)B_SZ * HID * 4);
  float* h1e             = (float*)alloc((size_t)B_SZ * HID * 4);
  float* c1e             = (float*)alloc((size_t)B_SZ * HID * 4);

  auto pack = [&](const float* s, unsigned short* d, int N, int K, int Kpad,
                  int stride, int so) {
    int total = N * Kpad;
    pack_weight_kernel<<<(total + 255) / 256, 256, 0, stream>>>(s, d, N, K, Kpad, stride, so);
  };
  pack(e_wih0, p_eih0, 1024, 16,  32,  16,  0);
  pack(e_whh0, p_ehh0, 1024, 256, 256, 256, 0);
  pack(e_wih1, p_eih1, 1024, 256, 256, 256, 0);
  pack(e_whh1, p_ehh1, 1024, 256, 256, 256, 0);
  pack(d_wih0, p_dih0, 1024, 257, 288, 257, 0);
  pack(d_whh0, p_dhh0, 1024, 256, 256, 256, 0);
  pack(d_wih1, p_dih1, 1024, 256, 256, 256, 0);
  pack(d_whh1, p_dhh1, 1024, 256, 256, 256, 0);
  pack(attn_w, p_we,   256,  256, 256, 512, 0);     // We = attn_w[:, :H]
  pack(attn_w, p_wd,   256,  256, 256, 512, 256);   // Wd = attn_w[:, H:]

  // Encoder layer 0: x = src (fp32, K=16 padded to 32)
  lstm_layer_kernel<32, true><<<B_SZ / 16, 256, 0, stream>>>(
      (const void*)src, p_eih0, p_ehh0, e_b0, y0, h0e, c0e);
  // Encoder layer 1: x = y0 (bf16, K=256) -- async staged
  lstm_layer_kernel<256, false><<<B_SZ / 16, 256, 0, stream>>>(
      (const void*)y0, p_eih1, p_ehh1, e_b1, eob, h1e, c1e);

  // enc_proj = enc_out @ We^T
  encproj_kernel<<<S_LEN * (B_SZ / 16), 512, 0, stream>>>(eob, p_we, epj);

  // Decoder
  decoder_kernel<<<B_SZ / 16, 256, 0, stream>>>(
      src, eob, epj, p_wd, attn_b, attn_v,
      p_dih0, p_dhh0, d_b0, p_dih1, p_dhh1, d_b1,
      fc_w, fc_b, h0e, c0e, h1e, c1e, out);

  (void)in_sizes; (void)n_in; (void)out_size; (void)ws_size;
}
